// Token_Purification_M_16527034155657
// MI455X (gfx1250) — matrix operand order repaired and checked by reading, not compiled, not run on hardware
//
#include <hip/hip_runtime.h>
#include <hip/hip_bf16.h>

// ---------- types ----------
typedef __attribute__((ext_vector_type(16))) __bf16 v16bf;
typedef __attribute__((ext_vector_type(8)))  float  v8f;

#define B_SZ   16
#define L_SZ   4096
#define D_SZ   256
#define H_SZ   4
#define HD_SZ  64
#define KEEP   2048
#define LP1    (L_SZ + 1)

__device__ __forceinline__ __bf16 f2bf(float f) {
  // round-to-nearest-even f32 -> bf16
  unsigned u = __builtin_bit_cast(unsigned, f);
  u += 0x7FFFu + ((u >> 16) & 1u);
  unsigned short h = (unsigned short)(u >> 16);
  return __builtin_bit_cast(__bf16, h);
}

// ---------------------------------------------------------------------------
// K1: q = score_emb @ Wq^T ; u[h,:] = sum_i q[h*64+i] * Wk[h*64+i,:] (scale
// 1/sqrt(64) folded in). Stored bf16, padded to 16 columns for the WMMA N dim.
// ---------------------------------------------------------------------------
__global__ void proj_u_kernel(const float* __restrict__ x_lead,
                              const float* __restrict__ Wq,
                              const float* __restrict__ Wk,
                              __bf16* __restrict__ u_pad) {
  __shared__ float s[D_SZ];
  __shared__ float q[D_SZ];
  const int b = blockIdx.x;
  const int d = threadIdx.x;          // 256 threads
  s[d] = x_lead[(size_t)b * LP1 * D_SZ + d];   // x_lead[b,0,:]
  __syncthreads();
  float acc = 0.f;
  #pragma unroll 8
  for (int i = 0; i < D_SZ; ++i) acc += s[i] * Wq[d * D_SZ + i];
  q[d] = acc;
  __syncthreads();
  const float scale = 0.125f;         // 64^-0.5
  for (int h = 0; h < H_SZ; ++h) {
    float u = 0.f;
    #pragma unroll 8
    for (int i = 0; i < HD_SZ; ++i)
      u += q[h * HD_SZ + i] * Wk[(size_t)(h * HD_SZ + i) * D_SZ + d];
    u_pad[(size_t)(b * 16 + h) * D_SZ + d] = f2bf(u * scale);
  }
  const __bf16 z = __builtin_bit_cast(__bf16, (unsigned short)0);
  for (int h = H_SZ; h < 16; ++h)
    u_pad[(size_t)(b * 16 + h) * D_SZ + d] = z;
}

// ---------------------------------------------------------------------------
// K2: dots[b,h,j] = u[b,h] . tokens[b,j]  via v_wmma_f32_16x16x32_bf16.
// One wave per 16-row token tile; K=256 -> 8 WMMAs. Memory-bound: reads the
// 268MB token tensor exactly once.
// ---------------------------------------------------------------------------
__global__ void dots_wmma_kernel(const float* __restrict__ x_lead,
                                 const __bf16* __restrict__ u_pad,
                                 float* __restrict__ dots) {
  const int lane = threadIdx.x & 31;
  const int wave = threadIdx.x >> 5;            // 8 waves per block
  const int tile = blockIdx.x * 8 + wave;       // 0 .. 4095
  const int b    = tile >> 8;                   // 256 tiles per batch
  const int row0 = (tile & 255) << 4;
  const int half = lane >> 4;                   // 0 / 1
  const int m    = lane & 15;                   // row (A) / col (B)

  const float*  rowp = x_lead + (size_t)b * LP1 * D_SZ + D_SZ   // skip score tok
                       + (size_t)(row0 + m) * D_SZ;
  const __bf16* up   = u_pad + (size_t)((b << 4) + m) * D_SZ + (half << 4);

  v8f c = {};
  #pragma unroll
  for (int kb = 0; kb < D_SZ; kb += 32) {
    // A fragment: 16x32 bf16. lanes<16 hold K kb+0..15,
    // lanes>=16 hold K kb+16..31 (matches B).
    const float* p = rowp + kb + half * 16;
    float4 f0 = *(const float4*)(p);
    float4 f1 = *(const float4*)(p + 4);
    float4 f2 = *(const float4*)(p + 8);
    float4 f3 = *(const float4*)(p + 12);
    v16bf a;
    a[0]  = f2bf(f0.x); a[1]  = f2bf(f0.y); a[2]  = f2bf(f0.z); a[3]  = f2bf(f0.w);
    a[4]  = f2bf(f1.x); a[5]  = f2bf(f1.y); a[6]  = f2bf(f1.z); a[7]  = f2bf(f1.w);
    a[8]  = f2bf(f2.x); a[9]  = f2bf(f2.y); a[10] = f2bf(f2.z); a[11] = f2bf(f2.w);
    a[12] = f2bf(f3.x); a[13] = f2bf(f3.y); a[14] = f2bf(f3.z); a[15] = f2bf(f3.w);
    // B fragment: 32x16 bf16. lane = column n, low half K kb+0..15,
    // high half K kb+16..31; contiguous 32B load.
    v16bf bb = *(const v16bf*)(up + kb);
    c = __builtin_amdgcn_wmma_f32_16x16x32_bf16(false, a, false, bb,
                                                (short)0, c, false, false);
  }
  // C layout: element r -> M = r + 8*half, N = lane&15. Only N<4 are heads.
  if (m < H_SZ) {
    float* dp = dots + ((size_t)(b * H_SZ + m) * L_SZ) + row0 + half * 8;
    float4 o0; o0.x = c[0]; o0.y = c[1]; o0.z = c[2]; o0.w = c[3];
    float4 o1; o1.x = c[4]; o1.y = c[5]; o1.z = c[6]; o1.w = c[7];
    *(float4*)(dp)     = o0;
    *(float4*)(dp + 4) = o1;
  }
}

// ---------------------------------------------------------------------------
// K3: per (b,h) softmax over L, then mean over heads -> scores[b,j].
// ---------------------------------------------------------------------------
__global__ void softmax_scores_kernel(const float* __restrict__ dots,
                                      float* __restrict__ scores) {
  const int b = blockIdx.x;
  const int t = threadIdx.x;          // 1024 threads
  __shared__ float red[1024];
  __shared__ float mx[H_SZ], inv[H_SZ];
  for (int h = 0; h < H_SZ; ++h) {
    const float* dp = dots + (size_t)(b * H_SZ + h) * L_SZ;
    float m = -3.402823466e38f;
    for (int j = t; j < L_SZ; j += 1024) m = fmaxf(m, dp[j]);
    red[t] = m; __syncthreads();
    for (int s = 512; s > 0; s >>= 1) {
      if (t < s) red[t] = fmaxf(red[t], red[t + s]);
      __syncthreads();
    }
    if (t == 0) mx[h] = red[0];
    __syncthreads();
    const float mh = mx[h];
    float sum = 0.f;
    for (int j = t; j < L_SZ; j += 1024) sum += __expf(dp[j] - mh);
    red[t] = sum; __syncthreads();
    for (int s = 512; s > 0; s >>= 1) {
      if (t < s) red[t] += red[t + s];
      __syncthreads();
    }
    if (t == 0) inv[h] = 1.0f / red[0];
    __syncthreads();
  }
  for (int j = t; j < L_SZ; j += 1024) {
    float sc = 0.f;
    #pragma unroll
    for (int h = 0; h < H_SZ; ++h)
      sc += __expf(dots[(size_t)(b * H_SZ + h) * L_SZ + j] - mx[h]) * inv[h];
    scores[(size_t)b * L_SZ + j] = sc * 0.25f;
  }
}

// ---------------------------------------------------------------------------
// K4: per-batch bitonic sort of 4096 keys = (orderable score desc, idx asc),
// emit first KEEP indices. 32KB LDS.
// ---------------------------------------------------------------------------
__global__ void topk_sort_kernel(const float* __restrict__ scores,
                                 int* __restrict__ ids) {
  const int b = blockIdx.x;
  const int t = threadIdx.x;          // 1024 threads
  __shared__ unsigned long long key[L_SZ];
  for (int j = t; j < L_SZ; j += 1024) {
    unsigned u = __builtin_bit_cast(unsigned, scores[(size_t)b * L_SZ + j]);
    u = (u & 0x80000000u) ? ~u : (u | 0x80000000u);   // monotonic map
    key[j] = ((unsigned long long)u << 32)
           | (unsigned long long)(0xFFFFFFFFu - (unsigned)j); // tiebreak: idx asc
  }
  __syncthreads();
  for (int k = 2; k <= L_SZ; k <<= 1) {
    for (int j = k >> 1; j > 0; j >>= 1) {
      for (int i = t; i < L_SZ; i += 1024) {
        const int l = i ^ j;
        if (l > i) {
          const bool down = ((i & k) == 0);   // descending blocks overall
          unsigned long long a = key[i], c = key[l];
          if (down ? (a < c) : (a > c)) { key[i] = c; key[l] = a; }
        }
      }
      __syncthreads();
    }
  }
  for (int i = t; i < KEEP; i += 1024)
    ids[b * KEEP + i] =
        (int)(0xFFFFFFFFu - (unsigned)(key[i] & 0xFFFFFFFFull));
}

// ---------------------------------------------------------------------------
// K5: gather x_sel and pos_sel (float4-coalesced).
// ---------------------------------------------------------------------------
__global__ void gather_kernel(const float* __restrict__ x_lead,
                              const float* __restrict__ x_pos,
                              const int* __restrict__ ids,
                              float* __restrict__ out) {
  const int gid = blockIdx.x * blockDim.x + threadIdx.x; // B*KEEP*64 threads
  const int c4  = gid & 63;
  const int i   = (gid >> 6) & (KEEP - 1);
  const int b   = gid >> 17;
  const int idx = ids[b * KEEP + i];
  const float4* st = (const float4*)(x_lead + (size_t)b * LP1 * D_SZ
                                     + (size_t)(1 + idx) * D_SZ) + c4;
  const float4* sp = (const float4*)(x_pos + ((size_t)b * L_SZ + idx) * D_SZ) + c4;
  float4* dx = (float4*)(out + (size_t)(b * KEEP + i) * D_SZ) + c4;
  float4* dp = (float4*)(out + (size_t)B_SZ * KEEP * D_SZ
                         + (size_t)(b * KEEP + i) * D_SZ) + c4;
  *dx = *st;
  *dp = *sp;
}

// ---------------------------------------------------------------------------
extern "C" void kernel_launch(void* const* d_in, const int* in_sizes, int n_in,
                              void* d_out, int out_size, void* d_ws, size_t ws_size,
                              hipStream_t stream) {
  (void)in_sizes; (void)n_in; (void)out_size; (void)ws_size;
  const float* x_lead = (const float*)d_in[0];
  const float* x_pos  = (const float*)d_in[1];
  const float* Wq     = (const float*)d_in[2];
  const float* Wk     = (const float*)d_in[3];
  float* out = (float*)d_out;

  char* ws = (char*)d_ws;
  __bf16* u_pad  = (__bf16*)ws;                                // 16*16*256*2  = 128KB
  float*  dots   = (float*)(ws + (1u << 17));                  // 16*4*4096*4  = 1MB
  float*  scores = (float*)(ws + (1u << 17) + (1u << 20));     // 16*4096*4    = 256KB
  int*    ids    = (int*)(ws + (1u << 17) + (1u << 20) + (1u << 18)); // 128KB

  proj_u_kernel      <<<B_SZ,            256, 0, stream>>>(x_lead, Wq, Wk, u_pad);
  dots_wmma_kernel   <<<(B_SZ*L_SZ/16)/8, 256, 0, stream>>>(x_lead, u_pad, dots);
  softmax_scores_kernel<<<B_SZ,         1024, 0, stream>>>(dots, scores);
  topk_sort_kernel   <<<B_SZ,           1024, 0, stream>>>(scores, ids);
  gather_kernel      <<<(B_SZ*KEEP*64)/256, 256, 0, stream>>>(x_lead, x_pos, ids, out);
}
